// StripedHyenaLayer_18665927868506
// MI455X (gfx1250) — matrix operand order repaired and checked by reading, not compiled, not run on hardware
//
#include <hip/hip_runtime.h>
#include <hip/hip_bf16.h>
#include <math.h>

// Problem dims
#define B_   8
#define L_   4096
#define D_   256
#define H_   768
#define M_   (B_ * L_)   // 32768 tokens
#define K_   256
#define NST  32

typedef __attribute__((ext_vector_type(16))) __bf16 v16bf;
typedef __attribute__((ext_vector_type(8)))  float  v8f;
typedef __attribute__((ext_vector_type(4)))  unsigned v4u;
typedef __attribute__((ext_vector_type(8)))  int      v8i;
typedef __attribute__((ext_vector_type(4)))  int      v4i;

struct U4 { unsigned x, y, z, w; };
struct U8 { U4 lo, hi; };
union AB32 { U8 u; v16bf v; };

#if __has_builtin(__builtin_amdgcn_tensor_load_to_lds) && __has_builtin(__builtin_amdgcn_s_wait_tensorcnt)
#define HAS_TDM 1
#else
#define HAS_TDM 0
#endif

__device__ __forceinline__ unsigned short f2bf(float f) {
    union { float f; unsigned u; } v; v.f = f;
    unsigned r = v.u + 0x7FFFu + ((v.u >> 16) & 1u);
    return (unsigned short)(r >> 16);
}

#if HAS_TDM
// TDM: DMA a contiguous 64x256 bf16 (32 KB) weight strip from global to LDS.
// Descriptor per ISA ch.8: count=1, data_size=2B, 1-row tile of 16384 elems, type=2.
__device__ __forceinline__ void tdm_load_strip_32k(const unsigned short* gsrc, void* ldst) {
    unsigned lds_lo = (unsigned)(size_t)ldst;
    unsigned long long ga = (unsigned long long)(size_t)gsrc;
    v4u g0 = { 1u, lds_lo, (unsigned)(ga & 0xFFFFFFFFu),
               (unsigned)((ga >> 32) & 0x1FFFFFFu) | (2u << 30) };
    v8i g1 = { (int)0x00010000,   // data_size=2B, no multicast
               (int)0x40000000,   // tensor_dim0 = 16384 (low16 in bits[31:16])
               (int)0x00010000,   // tensor_dim0 hi=0, tensor_dim1 = 1
               (int)0x40000000,   // tile_dim0 = 16384
               1,                 // tile_dim1 = 1
               16384,             // tensor_dim0_stride
               0, 0 };
    v4i g2 = { 0, 0, 0, 0 };
    v4i g3 = { 0, 0, 0, 0 };
#if __has_include(<hip/amd_detail/amd_gfx1250_TDM.h>)
    v8i g4 = { 0, 0, 0, 0, 0, 0, 0, 0 };
    __builtin_amdgcn_tensor_load_to_lds(g0, g1, g2, g3, g4, 0);
#else
    __builtin_amdgcn_tensor_load_to_lds(g0, g1, g2, g3, 0);
#endif
}
#endif

// ---------------- pack kernels ----------------

__global__ void cvt_bf16_kernel(const float* __restrict__ in,
                                unsigned short* __restrict__ out, int n) {
    int i = blockIdx.x * blockDim.x + threadIdx.x;
    int stride = gridDim.x * blockDim.x;
    for (; i < n; i += stride) out[i] = f2bf(in[i]);
}

// conv_w: [O=256][I=256][ksz]  ->  out: [tap][o][k] bf16
__global__ void pack_conv_kernel(const float* __restrict__ w,
                                 unsigned short* __restrict__ out, int ksz) {
    int n = D_ * D_ * ksz;
    int idx = blockIdx.x * blockDim.x + threadIdx.x;
    int stride = gridDim.x * blockDim.x;
    for (; idx < n; idx += stride) {
        int tap = idx / (D_ * D_);
        int rem = idx % (D_ * D_);
        int o = rem / D_;
        int k = rem % D_;
        out[idx] = f2bf(w[(o * D_ + k) * ksz + tap]);
    }
}

// ---------------- WMMA GEMM with TDM/LDS-staged weights ----------------
// out[t,o] = sum_k A[t,k]*W[o,k] + bias[o]
__global__ void __launch_bounds__(128)
wmma_gemm_kernel(const unsigned short* __restrict__ A,   // [M,K] bf16
                 const unsigned short* __restrict__ W,   // [N,K] bf16
                 const float* __restrict__ bias,         // [N]
                 float* __restrict__ Cout, int N) {
    __shared__ unsigned short wsh[64 * K_];   // 32 KB weight strip
    const int lane   = threadIdx.x & 31;
    const int wave   = threadIdx.x >> 5;
    const int mTile  = blockIdx.x * 4 + wave;
    const int nBase  = blockIdx.y * 64;
    const int mBase  = mTile * 16;
    const int khalf  = lane >> 4;
    const int row    = mBase + (lane & 15);
    const int colLoc = lane & 15;

#if HAS_TDM
    if (wave == 0) {
        tdm_load_strip_32k(W + (size_t)nBase * K_, &wsh[0]);
        __builtin_amdgcn_s_wait_tensorcnt(0);
    }
#else
    {
        const U4* src = reinterpret_cast<const U4*>(W + (size_t)nBase * K_);
        U4* dst = reinterpret_cast<U4*>(wsh);
        for (int i = threadIdx.x; i < (64 * K_) / 8; i += 128) dst[i] = src[i];
    }
#endif

    // Preload the full K=256 A strip for this wave (8 fragments, issued as clauses)
    const unsigned short* arow = A + (size_t)row * K_ + khalf * 16;
    AB32 af[8];
#pragma unroll
    for (int kk = 0; kk < 8; ++kk) {
        af[kk].u.lo = *reinterpret_cast<const U4*>(arow + kk * 32);
        af[kk].u.hi = *reinterpret_cast<const U4*>(arow + kk * 32 + 8);
    }

    __syncthreads();   // weights visible in LDS

    v8f acc[4] = {v8f{}, v8f{}, v8f{}, v8f{}};
#pragma unroll
    for (int kk = 0; kk < 8; ++kk) {
        const int k0 = kk * 32;
        AB32 bf[4];
#pragma unroll
        for (int j = 0; j < 4; ++j) {
            const unsigned short* wl =
                wsh + (size_t)(colLoc + j * 16) * K_ + k0 + khalf * 16;
            bf[j].u.lo = *reinterpret_cast<const U4*>(wl);
            bf[j].u.hi = *reinterpret_cast<const U4*>(wl + 8);
        }
#pragma unroll
        for (int j = 0; j < 4; ++j)
            acc[j] = __builtin_amdgcn_wmma_f32_16x16x32_bf16(
                false, af[kk].v, false, bf[j].v, (short)0, acc[j], false, false);
    }

    const int mrow = khalf * 8;   // C/D layout: lanes>=16 hold rows M=8..15
#pragma unroll
    for (int j = 0; j < 4; ++j) {
        int c = nBase + colLoc + j * 16;
        float bv = bias[c];
#pragma unroll
        for (int v = 0; v < 8; ++v)
            Cout[(size_t)(mBase + mrow + v) * N + c] = acc[j][v] + bv;
    }
}

// ---------------- WMMA dilated conv: shifted tap-GEMMs, 16x64 tile/wave ----------------
// Per-tap 32 KB weight strips are double-buffered through LDS (TDM ping-pong:
// issue tap+1 while computing tap; TENSORcnt is in-order so wait<=1 suffices).
template <int KSZ, int DIL>
__device__ __forceinline__ void conv_body(const unsigned short* __restrict__ A,
                                          const unsigned short* __restrict__ Wp,
                                          const float* __restrict__ bias,
                                          float* __restrict__ Cout,
                                          unsigned short* wtile,   // LDS [2][64*K_]
                                          int mBase, int oBase, int tapBase, int branch) {
    const int lane   = threadIdx.x & 31;
    const int wave   = threadIdx.x >> 5;
    const int khalf  = lane >> 4;
    const int colLoc = lane & 15;
    const int tloc   = (mBase & (L_ - 1)) + colLoc;        // token-in-batch of lane's row
    const size_t batchBase = (size_t)(mBase & ~(L_ - 1));
    const unsigned short* wstrip = Wp + (size_t)tapBase * (D_ * D_) + (size_t)oBase * K_;

    v8f acc[4] = {v8f{}, v8f{}, v8f{}, v8f{}};

#if HAS_TDM
    if (wave == 0) tdm_load_strip_32k(wstrip, wtile);                 // tap 0 -> buf 0
#endif

#pragma unroll
    for (int tap = 0; tap < KSZ; ++tap) {
        unsigned short* buf = wtile + (size_t)(tap & 1) * (64 * K_);

        // A fragments for this tap (global clause, overlaps the DMA)
        const int off = (tap - (KSZ >> 1)) * DIL;
        const int t = tloc + off;
        const bool valid = (t >= 0) && (t < L_);
        const unsigned short* arow = A + (batchBase + (size_t)t) * K_ + khalf * 16;
        __builtin_prefetch(arow, 0, 1);
        AB32 af[8];
#pragma unroll
        for (int kk = 0; kk < 8; ++kk) {
            if (valid) {
                af[kk].u.lo = *reinterpret_cast<const U4*>(arow + kk * 32);
                af[kk].u.hi = *reinterpret_cast<const U4*>(arow + kk * 32 + 8);
            } else {
                af[kk].u.lo = U4{0u, 0u, 0u, 0u};
                af[kk].u.hi = U4{0u, 0u, 0u, 0u};
            }
        }

#if HAS_TDM
        if (wave == 0) {
            if (tap + 1 < KSZ) {   // prefetch next tap into the other buffer
                tdm_load_strip_32k(wstrip + (size_t)(tap + 1) * (D_ * D_),
                                   wtile + (size_t)((tap + 1) & 1) * (64 * K_));
                __builtin_amdgcn_s_wait_tensorcnt(1);   // tap's DMA done (in-order)
            } else {
                __builtin_amdgcn_s_wait_tensorcnt(0);   // last tap: drain
            }
        }
#else
        {
            const U4* src = reinterpret_cast<const U4*>(wstrip + (size_t)tap * (D_ * D_));
            U4* dst = reinterpret_cast<U4*>(buf);
            for (int i = threadIdx.x; i < (64 * K_) / 8; i += 128) dst[i] = src[i];
        }
#endif
        __syncthreads();   // strip for this tap visible in LDS

#pragma unroll
        for (int kk = 0; kk < 8; ++kk) {
            const int k0 = kk * 32;
            AB32 bfr[4];
#pragma unroll
            for (int j = 0; j < 4; ++j) {
                const unsigned short* wl =
                    buf + (size_t)(j * 16 + colLoc) * K_ + k0 + khalf * 16;
                bfr[j].u.lo = *reinterpret_cast<const U4*>(wl);
                bfr[j].u.hi = *reinterpret_cast<const U4*>(wl + 8);
            }
#pragma unroll
            for (int j = 0; j < 4; ++j)
                acc[j] = __builtin_amdgcn_wmma_f32_16x16x32_bf16(
                    false, af[kk].v, false, bfr[j].v, (short)0, acc[j], false, false);
        }
        __syncthreads();   // all waves done with this buffer before it is refilled
    }

    const int mrow = khalf * 8;
#pragma unroll
    for (int j = 0; j < 4; ++j) {
        const int oloc = oBase + j * 16 + colLoc;       // branch-local channel
        const float bv = bias[oloc];
        const int oc = branch * D_ + oloc;              // concat channel
#pragma unroll
        for (int v = 0; v < 8; ++v)
            Cout[(size_t)(mBase + mrow + v) * H_ + oc] = acc[j][v] + bv;
    }
}

__global__ void __launch_bounds__(128)
wmma_conv_kernel(const unsigned short* __restrict__ A,    // [M,K] bf16 tokens
                 const unsigned short* __restrict__ Wp,   // packed [25][256][256] bf16
                 const float* __restrict__ b0, const float* __restrict__ b1,
                 const float* __restrict__ b2,
                 float* __restrict__ Cout) {               // [M,H]
    __shared__ unsigned short wtile[2][64 * K_];   // 64 KB ping-pong weight strips
    const int wave    = threadIdx.x >> 5;
    const int mBase   = (blockIdx.x * 4 + wave) * 16;
    const int nTile64 = blockIdx.y;            // 0..11
    const int branch  = nTile64 >> 2;          // 4 tiles of 64 per 256-ch branch
    const int oBase   = (nTile64 & 3) * 64;    // branch-local column base

    if (branch == 0)      conv_body<3, 1>(A, Wp, b0, Cout, &wtile[0][0], mBase, oBase, 0,  0);
    else if (branch == 1) conv_body<7, 2>(A, Wp, b1, Cout, &wtile[0][0], mBase, oBase, 3,  1);
    else                  conv_body<15, 4>(A, Wp, b2, Cout, &wtile[0][0], mBase, oBase, 10, 2);
}

// ---------------- gate*sigmoid + transpose [token,H] -> [B,H,L] ----------------
__global__ void __launch_bounds__(256)
combine_kernel(const float* __restrict__ conv, const float* __restrict__ gate,
               float* __restrict__ out) {
    __shared__ float tile[32][33];
    const int t0 = blockIdx.x * 32;
    const int h0 = blockIdx.y * 32;
    const int b  = blockIdx.z;
    const int tx = threadIdx.x & 31;
    const int ty = threadIdx.x >> 5;   // 0..7
#pragma unroll
    for (int r = 0; r < 4; ++r) {
        int trow = ty + r * 8;
        size_t idx = ((size_t)(b * L_ + t0 + trow)) * H_ + h0 + tx;
        float g = gate[idx];
        float s = 1.f / (1.f + __expf(-g));
        tile[trow][tx] = conv[idx] * s;
    }
    __syncthreads();
#pragma unroll
    for (int r = 0; r < 4; ++r) {
        int hrow = ty + r * 8;
        out[((size_t)(b * H_ + h0 + hrow)) * L_ + t0 + tx] = tile[tx][hrow];
    }
}

// ---------------- S4D diagonal-state scan (ZOH), wave per (b,h) ----------------
// y[t] = 2*Re( sum_n C_disc[n] * x_n[t] ),  x_n[t] = w_n x_n[t-1] + u[t]
// Identical to the causal FFT convolution of the reference.
__global__ void __launch_bounds__(256)
s4d_scan_kernel(const float* __restrict__ u,         // [B,H,L]
                const float* __restrict__ log_dt,
                const float* __restrict__ A_re, const float* __restrict__ A_im,
                const float* __restrict__ C_re, const float* __restrict__ C_im,
                const float* __restrict__ Dskip,
                float* __restrict__ y) {             // [token,H]
    __shared__ float ush[8][32];
    __shared__ float mat[8][32][33];
    const int lane = threadIdx.x & 31;   // state index n
    const int wv   = threadIdx.x >> 5;
    const int wh   = blockIdx.x * 8 + wv;
    const int b = wh / H_;
    const int h = wh % H_;

    const float dt = __expf(log_dt[h]);
    const float ar = A_re[h * NST + lane], ai = A_im[h * NST + lane];
    const float cr = C_re[h * NST + lane], ci = C_im[h * NST + lane];
    const float dr = dt * ar, di = dt * ai;
    const float er = __expf(dr);
    float sn, cs; __sincosf(di, &sn, &cs);
    const float wr = er * cs, wi = er * sn;       // w = exp(dt*A)
    const float emr = wr - 1.f, emi = wi;         // exp(dt*A) - 1
    const float den = ar * ar + ai * ai;
    const float qr = (emr * ar + emi * ai) / den; // (e-1)/A
    const float qi = (emi * ar - emr * ai) / den;
    const float cdr = 2.f * (cr * qr - ci * qi);  // 2*C_disc
    const float cdi = 2.f * (cr * qi + ci * qr);
    const float dk = Dskip[h];

    const float* ubase = u + ((size_t)(b * H_ + h)) * L_;
    float xr = 0.f, xi = 0.f;

    for (int c = 0; c < L_ / 32; ++c) {
        float uv = ubase[c * 32 + lane];
        ush[wv][lane] = uv;
        __syncthreads();
        for (int t = 0; t < 32; ++t) {
            float ut  = ush[wv][t];              // LDS broadcast
            float nxr = wr * xr - wi * xi + ut;
            float nxi = wr * xi + wi * xr;
            xr = nxr; xi = nxi;
            mat[wv][lane][t] = cdr * xr - cdi * xi;  // Re(2C*x) contribution
        }
        __syncthreads();
        float s = 0.f;
#pragma unroll
        for (int nn = 0; nn < 32; ++nn) s += mat[wv][nn][lane];
        s += dk * ush[wv][lane];
        int token = b * L_ + c * 32 + lane;
        y[(size_t)token * H_ + h] = s;
        __syncthreads();
    }
}

// ---------------- residual add + LayerNorm over H=768, block per token ----------------
__global__ void __launch_bounds__(256)
final_ln_kernel(const float* __restrict__ y, const float* __restrict__ res,
                const float* __restrict__ gamma, const float* __restrict__ beta,
                float* __restrict__ out) {
    __shared__ float red[256];
    __shared__ float red2[256];
    const int token = blockIdx.x;
    const int tid = threadIdx.x;
    const size_t base = (size_t)token * H_;
    float v0 = y[base + tid]       + res[base + tid];
    float v1 = y[base + tid + 256] + res[base + tid + 256];
    float v2 = y[base + tid + 512] + res[base + tid + 512];
    red[tid]  = v0 + v1 + v2;
    red2[tid] = v0 * v0 + v1 * v1 + v2 * v2;
    __syncthreads();
    for (int w = 128; w > 0; w >>= 1) {
        if (tid < w) { red[tid] += red[tid + w]; red2[tid] += red2[tid + w]; }
        __syncthreads();
    }
    float mean = red[0] * (1.f / H_);
    float var  = red2[0] * (1.f / H_) - mean * mean;
    float inv  = rsqrtf(var + 1e-5f);
    out[base + tid]       = (v0 - mean) * inv * gamma[tid]       + beta[tid];
    out[base + tid + 256] = (v1 - mean) * inv * gamma[tid + 256] + beta[tid + 256];
    out[base + tid + 512] = (v2 - mean) * inv * gamma[tid + 512] + beta[tid + 512];
}

// ---------------- launch ----------------
extern "C" void kernel_launch(void* const* d_in, const int* in_sizes, int n_in,
                              void* d_out, int out_size, void* d_ws, size_t ws_size,
                              hipStream_t stream) {
    (void)in_sizes; (void)n_in; (void)out_size; (void)ws_size;
    const float* x    = (const float*)d_in[0];
    const float* cw0  = (const float*)d_in[1];
    const float* cb0  = (const float*)d_in[2];
    const float* cw1  = (const float*)d_in[3];
    const float* cb1  = (const float*)d_in[4];
    const float* cw2  = (const float*)d_in[5];
    const float* cb2  = (const float*)d_in[6];
    const float* gw   = (const float*)d_in[7];
    const float* gb   = (const float*)d_in[8];
    const float* rw   = (const float*)d_in[9];
    const float* rb   = (const float*)d_in[10];
    const float* ldt  = (const float*)d_in[11];
    const float* Are  = (const float*)d_in[12];
    const float* Aim  = (const float*)d_in[13];
    const float* Cre  = (const float*)d_in[14];
    const float* Cim  = (const float*)d_in[15];
    const float* Dk   = (const float*)d_in[16];
    const float* gam  = (const float*)d_in[17];
    const float* bet  = (const float*)d_in[18];
    float* out = (float*)d_out;

    char* ws = (char*)d_ws;
    size_t o = 0;
    unsigned short* a_bf  = (unsigned short*)(ws + o); o += (size_t)M_ * K_ * 2;
    unsigned short* wg_bf = (unsigned short*)(ws + o); o += (size_t)H_ * K_ * 2;
    unsigned short* wr_bf = (unsigned short*)(ws + o); o += (size_t)H_ * K_ * 2;
    unsigned short* wc_bf = (unsigned short*)(ws + o); o += (size_t)25 * D_ * D_ * 2;
    float* gate_lin = (float*)(ws + o); o += (size_t)M_ * H_ * 4;
    float* res_lin  = (float*)(ws + o); o += (size_t)M_ * H_ * 4;
    float* conv_out = (float*)(ws + o); o += (size_t)M_ * H_ * 4;
    float* gated    = (float*)(ws + o); o += (size_t)M_ * H_ * 4;
    float* y_tr = conv_out;   // conv_out dead after combine_kernel; reuse for scan output

    cvt_bf16_kernel<<<2048, 256, 0, stream>>>(x,  a_bf,  M_ * K_);
    cvt_bf16_kernel<<<192,  256, 0, stream>>>(gw, wg_bf, H_ * K_);
    cvt_bf16_kernel<<<192,  256, 0, stream>>>(rw, wr_bf, H_ * K_);
    pack_conv_kernel<<<768,  256, 0, stream>>>(cw0, wc_bf,                3);
    pack_conv_kernel<<<1792, 256, 0, stream>>>(cw1, wc_bf + 3  * D_ * D_, 7);
    pack_conv_kernel<<<3840, 256, 0, stream>>>(cw2, wc_bf + 10 * D_ * D_, 15);

    dim3 ggrid(M_ / 64, H_ / 64);   // 512 x 12
    wmma_gemm_kernel<<<ggrid, 128, 0, stream>>>(a_bf, wg_bf, gb, gate_lin, H_);
    wmma_gemm_kernel<<<ggrid, 128, 0, stream>>>(a_bf, wr_bf, rb, res_lin,  H_);

    dim3 cgrid(M_ / 64, 12);        // 512 x 12, 16x64 tile per wave
    wmma_conv_kernel<<<cgrid, 128, 0, stream>>>(a_bf, wc_bf, cb0, cb1, cb2, conv_out);

    dim3 tgrid(L_ / 32, H_ / 32, B_);  // 128 x 24 x 8
    combine_kernel<<<tgrid, 256, 0, stream>>>(conv_out, gate_lin, gated);

    s4d_scan_kernel<<<(B_ * H_) / 8, 256, 0, stream>>>(gated, ldt, Are, Aim,
                                                       Cre, Cim, Dk, y_tr);

    final_ln_kernel<<<M_, 256, 0, stream>>>(y_tr, res_lin, gam, bet, out);
}
